// MultiScaleTriplane_46102178955453
// MI455X (gfx1250) — matrix operand order repaired and checked by reading, not compiled, not run on hardware
//
#include <hip/hip_runtime.h>
#include <hip/hip_bf16.h>

// ---------------------------------------------------------------------------
// MultiScaleTriplane for MI455X (gfx1250, wave32, WMMA)
//   prep : transpose (3,C,H,W) -> (3,H,W,C) into d_ws so lane==channel gathers
//          are 128B-coalesced and L2-resident (params total ~31MB << 192MB L2)
//   main : per-block 128 points; bicubic/cubic features -> LDS bf16 A-tile,
//          folded-B' (K=96) transposed bf16 in LDS, v_wmma_f32_16x16x32_bf16,
//          v_sin/v_cos (hardware computes sin(2*pi*x), matching the reference
//          2*pi scale for free), coalesced unguarded stores on full tiles.
// ---------------------------------------------------------------------------

typedef __attribute__((ext_vector_type(16))) __bf16 v16bf;
typedef __attribute__((ext_vector_type(8)))  float  v8f;

#define NPTS 128   // points per block
#define KF   96    // folded feature length (f3 | f2 | fv)
#define AST  104   // padded LDS row stride (bf16 elems): 208B, 16B-aligned rows

__device__ __forceinline__ unsigned short f2bf(float f) {
  unsigned u = __float_as_uint(f);
  u += 0x7FFFu + ((u >> 16) & 1u);   // round-to-nearest-even
  return (unsigned short)(u >> 16);
}

struct Taps { float w[4]; int i[4]; };

// torch bicubic (A=-0.75), align_corners=True, border clamp
__device__ __forceinline__ Taps axis_taps(float coord, int size) {
  const float A = -0.75f;
  float ix = (coord + 1.0f) * 0.5f * (float)(size - 1);
  float x0 = floorf(ix);
  float t  = ix - x0;
  Taps tp;
  float x;
  x = t + 1.0f; tp.w[0] = ((A * x - 5.0f * A) * x + 8.0f * A) * x - 4.0f * A;
  x = t;        tp.w[1] = ((A + 2.0f) * x - (A + 3.0f)) * x * x + 1.0f;
  x = 1.0f - t; tp.w[2] = ((A + 2.0f) * x - (A + 3.0f)) * x * x + 1.0f;
  x = 2.0f - t; tp.w[3] = ((A * x - 5.0f * A) * x + 8.0f * A) * x - 4.0f * A;
  int b = (int)x0;
#pragma unroll
  for (int k = 0; k < 4; ++k) {
    int id = b - 1 + k;
    id = id < 0 ? 0 : id;
    id = id > size - 1 ? size - 1 : id;
    tp.i[k] = id;
  }
  return tp;
}

// P is (H,W,32) float, channel-last; c = lane
__device__ __forceinline__ float sample_plane_c(const float* __restrict__ P, int W,
                                                const Taps& tx, const Taps& ty, int c) {
  float acc = 0.0f;
#pragma unroll
  for (int i = 0; i < 4; ++i) {
    const float* row = P + ((size_t)ty.i[i] * (size_t)W) * 32u;
    float p = 0.0f;
#pragma unroll
    for (int j = 0; j < 4; ++j)
      p = fmaf(tx.w[j], row[(size_t)tx.i[j] * 32u + c], p);
    acc = fmaf(ty.w[i], p, acc);
  }
  return acc;
}

// (P,C,H,W) -> (P,H,W,C), C==32
__global__ void t_chw_hwc(const float* __restrict__ in, float* __restrict__ out,
                          int H, int W, long total) {
  long i = (long)blockIdx.x * blockDim.x + threadIdx.x;
  if (i >= total) return;
  int  c = (int)(i & 31);
  long r = i >> 5;
  int  w = (int)(r % W); r /= W;
  int  h = (int)(r % H); r /= H;
  long pl = r;
  out[i] = in[(((pl * 32 + c) * (long)H + h) * (long)W) + w];
}

__device__ __forceinline__ v16bf ld_frag(const unsigned short* p) {
  union { uint4 u[2]; v16bf v; } r;
  r.u[0] = *(const uint4*)(p);       // K 0..7  (or 8..15 on high lanes)
  r.u[1] = *(const uint4*)(p + 16);  // K 16..23 (or 24..31)
  return r.v;
}

__global__ __launch_bounds__(256) void triplane_main(
    const float* __restrict__ coords,
    const float* __restrict__ p3t,   // (3,128,128,32)
    const float* __restrict__ p2t,   // (3,256,256,32)
    const float* __restrict__ vt,    // (3,512,32)
    const float* __restrict__ B,     // (128,64) row-major
    float* __restrict__ out,         // (N,128)
    int Ntot) {
  __shared__ __align__(16) unsigned short Atile[NPTS * AST];
  __shared__ __align__(16) unsigned short Bt[64 * AST];

  const int tid  = threadIdx.x;
  const int lane = tid & 31;
  const int wave = tid >> 5;
  const int pbase = blockIdx.x * NPTS;

  // ---- build folded B'^T in LDS (bf16): Bt[n][k], k<32: B[k]+B[k+32], else B[k+32]
  for (int idx = tid; idx < 64 * KF; idx += 256) {
    int n = idx / KF, k = idx - n * KF;
    float v = (k < 32) ? (B[k * 64 + n] + B[(k + 32) * 64 + n])
                       : B[(k + 32) * 64 + n];
    Bt[n * AST + k] = f2bf(v);
  }

  // ---- phase 1: features (lane == channel), 16 points per wave
  for (int pp = 0; pp < 16; ++pp) {
    int p  = wave * 16 + pp;
    int gp = pbase + p;
    if (gp > Ntot - 1) gp = Ntot - 1;
    float cx = coords[(size_t)gp * 3 + 0];
    float cy = coords[(size_t)gp * 3 + 1];
    float cz = coords[(size_t)gp * 3 + 2];

    Taps tx3 = axis_taps(cx, 128), ty3 = axis_taps(cy, 128), tz3 = axis_taps(cz, 128);
    const size_t S3 = (size_t)128 * 128 * 32;
    float f3 = sample_plane_c(p3t,          128, tx3, ty3, lane)   // (x,y)
             + sample_plane_c(p3t + S3,     128, ty3, tz3, lane)   // (y,z)
             + sample_plane_c(p3t + 2 * S3, 128, tx3, tz3, lane);  // (x,z)

    Taps tx2 = axis_taps(cx, 256), ty2 = axis_taps(cy, 256), tz2 = axis_taps(cz, 256);
    const size_t S2 = (size_t)256 * 256 * 32;
    float f2 = sample_plane_c(p2t,          256, tx2, ty2, lane)
             + sample_plane_c(p2t + S2,     256, ty2, tz2, lane)
             + sample_plane_c(p2t + 2 * S2, 256, tx2, tz2, lane);

    Taps vx = axis_taps(cx, 512), vy = axis_taps(cy, 512), vz = axis_taps(cz, 512);
    float fv = 0.0f;
#pragma unroll
    for (int i = 0; i < 4; ++i) {
      fv = fmaf(vx.w[i], vt[((size_t)0 * 512 + vx.i[i]) * 32 + lane], fv);
      fv = fmaf(vy.w[i], vt[((size_t)1 * 512 + vy.i[i]) * 32 + lane], fv);
      fv = fmaf(vz.w[i], vt[((size_t)2 * 512 + vz.i[i]) * 32 + lane], fv);
    }

    Atile[p * AST + lane]      = f2bf(f3);
    Atile[p * AST + 32 + lane] = f2bf(f2);
    Atile[p * AST + 64 + lane] = f2bf(fv);
  }

  __syncthreads();

  // ---- phase 2: (16x96)@(96x64) per wave via v_wmma_f32_16x16x32_bf16
  const int lm = lane & 15;
  const int kh = (lane >> 4) * 8;          // low lanes K 0..7, high lanes K 8..15
  const int mhi = (lane >> 4) * 8;         // D rows: v (lanes 0-15) / v+8 (lanes 16-31)
  const unsigned short* arow = &Atile[(wave * 16 + lm) * AST];
  const bool full = (pbase + NPTS <= Ntot);      // uniform across the block

#pragma unroll
  for (int nt = 0; nt < 4; ++nt) {
    v8f acc = {};
    const unsigned short* brow = &Bt[(nt * 16 + lm) * AST];
#pragma unroll
    for (int kt = 0; kt < 3; ++kt) {
      int bk = kt * 32 + kh;
      v16bf a = ld_frag(arow + bk);
      v16bf b = ld_frag(brow + bk);
      acc = __builtin_amdgcn_wmma_f32_16x16x32_bf16(
          false, a, false, b, (short)0, acc, false, false);
    }
    // D layout: VGPR v -> row v (lanes 0-15) / v+8 (lanes 16-31); col = lane&15
    const int col = nt * 16 + lm;
    float* orow = out + (size_t)(pbase + wave * 16 + mhi) * 128 + col;
    if (full) {
      // straight-line, clause-friendly: 16 coalesced b32 stores per tile
#pragma unroll
      for (int v = 0; v < 8; ++v) {
        float pr = acc[v];
        // v_sin_f32(x) == sin(2*pi*x): reference's 2*pi scale is implicit
        orow[(size_t)v * 128]      = __builtin_amdgcn_sinf(pr);
        orow[(size_t)v * 128 + 64] = __builtin_amdgcn_cosf(pr);
      }
    } else {
#pragma unroll
      for (int v = 0; v < 8; ++v) {
        int m = pbase + wave * 16 + mhi + v;
        if (m < Ntot) {
          float pr = acc[v];
          orow[(size_t)v * 128]      = __builtin_amdgcn_sinf(pr);
          orow[(size_t)v * 128 + 64] = __builtin_amdgcn_cosf(pr);
        }
      }
    }
  }
}

extern "C" void kernel_launch(void* const* d_in, const int* in_sizes, int n_in,
                              void* d_out, int out_size, void* d_ws, size_t ws_size,
                              hipStream_t stream) {
  const float* coords = (const float*)d_in[0];   // (N,3)
  const float* plane3 = (const float*)d_in[1];   // (3,32,128,128)
  const float* plane2 = (const float*)d_in[2];   // (3,32,256,256)
  const float* vec1   = (const float*)d_in[3];   // (3,32,512)
  const float* Bm     = (const float*)d_in[4];   // (128,64)
  float* out = (float*)d_out;

  const long n3 = 3L * 32 * 128 * 128;
  const long n2 = 3L * 32 * 256 * 256;
  const long nv = 3L * 32 * 512;

  float* p3t = (float*)d_ws;
  float* p2t = p3t + n3;
  float* vt  = p2t + n2;

  t_chw_hwc<<<(int)((n3 + 255) / 256), 256, 0, stream>>>(plane3, p3t, 128, 128, n3);
  t_chw_hwc<<<(int)((n2 + 255) / 256), 256, 0, stream>>>(plane2, p2t, 256, 256, n2);
  t_chw_hwc<<<(int)((nv + 255) / 256), 256, 0, stream>>>(vec1,   vt,  1,   512, nv);

  int N = in_sizes[0] / 3;
  int blocks = (N + NPTS - 1) / NPTS;
  triplane_main<<<blocks, 256, 0, stream>>>(coords, p3t, p2t, vt, Bm, out, N);
}
